// OpenaiMLP_18313740550804
// MI455X (gfx1250) — matrix operand order repaired and checked by reading, not compiled, not run on hardware
//
#include <hip/hip_runtime.h>
#include <math.h>
#include <stdint.h>

#define T_TOK 2048
#define HDIM  2048
#define IDIM  2880
#define NEXP  8

typedef __attribute__((ext_vector_type(16))) __bf16 v16bf;
typedef __attribute__((ext_vector_type(8)))  float  v8f;

union FragU { uint4 q[2]; v16bf v; };

static __device__ __forceinline__ unsigned short f2bf(float f) {
  unsigned int u = __float_as_uint(f);
  u += 0x7FFFu + ((u >> 16) & 1u);          // round-to-nearest-even
  return (unsigned short)(u >> 16);
}
static __device__ __forceinline__ unsigned int pk2(float a, float b) {
  return (unsigned int)f2bf(a) | ((unsigned int)f2bf(b) << 16);
}
// low 32 bits of a generic pointer to __shared__ == LDS byte offset (flat->LDS truncation)
static __device__ __forceinline__ unsigned lds_off32(const void* p) {
  return (unsigned)(unsigned long long)(uintptr_t)p;
}
// CDNA5 async global->LDS copy, 16B per lane, tracked with ASYNCcnt
static __device__ __forceinline__ void async_b128(unsigned lds_dst, const void* gsrc) {
  asm volatile("global_load_async_to_lds_b128 %0, %1, off"
               :: "v"(lds_dst), "v"((unsigned long long)(uintptr_t)gsrc)
               : "memory");
}
static __device__ __forceinline__ void wait_async_le4() {
  asm volatile("s_wait_asynccnt 0x4" ::: "memory");
}
static __device__ __forceinline__ void wait_async_0() {
  asm volatile("s_wait_asynccnt 0x0" ::: "memory");
}

// ---------------------------------------------------------------- router
__global__ void moe_router_kernel(const float* __restrict__ x,
                                  const float* __restrict__ rw,
                                  float* __restrict__ scores,     // [E][T]
                                  int* __restrict__ slot_token,
                                  float* __restrict__ slot_w,
                                  int* __restrict__ slot_k,
                                  int* __restrict__ counts) {
  const int t = blockIdx.x * blockDim.x + threadIdx.x;
  if (t >= T_TOK) return;
  float acc[NEXP];
#pragma unroll
  for (int e = 0; e < NEXP; ++e) acc[e] = 0.f;
  const float* xr = x + (size_t)t * HDIM;
  for (int h = 0; h < HDIM; h += 4) {
    const float4 xv = *(const float4*)(xr + h);
#pragma unroll
    for (int e = 0; e < NEXP; ++e) {
      const float4 wv = *(const float4*)(rw + (size_t)e * HDIM + h);
      acc[e] += xv.x * wv.x + xv.y * wv.y + xv.z * wv.z + xv.w * wv.w;
    }
  }
  int i0 = 0; float v0 = acc[0];
#pragma unroll
  for (int e = 1; e < NEXP; ++e) if (acc[e] > v0) { v0 = acc[e]; i0 = e; }
  int i1 = -1; float v1 = -INFINITY;
#pragma unroll
  for (int e = 0; e < NEXP; ++e) if (e != i0 && acc[e] > v1) { v1 = acc[e]; i1 = e; }
  const float s0 = 1.f / (1.f + __expf(-v0));
  const float s1 = 1.f / (1.f + __expf(-v1));
#pragma unroll
  for (int e = 0; e < NEXP; ++e) scores[(size_t)e * T_TOK + t] = 0.f;
  scores[(size_t)i0 * T_TOK + t] = s0;
  scores[(size_t)i1 * T_TOK + t] = s1;
  int p = atomicAdd(&counts[i0], 1);
  int s = i0 * T_TOK + p;
  slot_token[s] = t; slot_w[s] = s0; slot_k[s] = 0;
  p = atomicAdd(&counts[i1], 1);
  s = i1 * T_TOK + p;
  slot_token[s] = t; slot_w[s] = s1; slot_k[s] = 1;
}

__global__ void moe_zero_counts(int* c) { if (threadIdx.x < NEXP) c[threadIdx.x] = 0; }

// ------------------------------------------------- fp32 -> bf16 (copy layout)
__global__ void moe_conv_kernel(const float* __restrict__ in,
                                unsigned short* __restrict__ out, int n) {
  const int i = blockIdx.x * blockDim.x + threadIdx.x;
  if (i < n) out[i] = f2bf(in[i]);
}

// ------------------------------- fp32 [b][R][C] -> bf16 [b][C][R] (transpose)
__global__ void moe_transpose_conv(const float* __restrict__ in,
                                   unsigned short* __restrict__ out,
                                   int R, int C) {
  __shared__ float tile[32][33];
  const int b  = blockIdx.z;
  const int c0 = blockIdx.x * 32;
  const int r0 = blockIdx.y * 32;
  const float* inb = in + (size_t)b * R * C;
  unsigned short* outb = out + (size_t)b * R * C;
  const int tx = threadIdx.x, ty = threadIdx.y;   // 32 x 8
#pragma unroll
  for (int i = 0; i < 32; i += 8) {
    const int r = r0 + ty + i, c = c0 + tx;
    tile[ty + i][tx] = (r < R && c < C) ? inb[(size_t)r * C + c] : 0.f;
  }
  __syncthreads();
#pragma unroll
  for (int i = 0; i < 32; i += 8) {
    const int c = c0 + ty + i, r = r0 + tx;
    if (r < R && c < C) outb[(size_t)c * R + r] = f2bf(tile[tx][ty + i]);
  }
}

// -------------------------------------------------------------- gate_up GEMM
// C = SiLU(A Wg^T) * (A Wu^T), A from x (f32, optional gather+scale), K = HDIM
template <bool GATHER>
__global__ __launch_bounds__(256)
void moe_gateup_gemm(const float* __restrict__ x,
                     const unsigned short* __restrict__ wg_,
                     const unsigned short* __restrict__ wu_,
                     unsigned long long westride,
                     unsigned short* __restrict__ act,      // rows e*T + m, [IDIM]
                     const int* __restrict__ slot_token,
                     const float* __restrict__ slot_w,
                     const int* __restrict__ counts) {
  const int e  = GATHER ? (int)blockIdx.z : 0;
  const int M  = GATHER ? counts[e] : T_TOK;
  const int m0 = blockIdx.y * 128;
  if (m0 >= M) return;
  const int n0 = blockIdx.x * 128;
  const unsigned short* wg = wg_ + (size_t)e * westride;
  const unsigned short* wu = wu_ + (size_t)e * westride;

  __shared__ unsigned short lds_a[2][128][32];
  __shared__ unsigned short lds_g[2][128][32];
  __shared__ unsigned short lds_u[2][128][32];
  __shared__ uint4 lds_trash[64];

  const int tid = threadIdx.x, lane = tid & 31, wave = tid >> 5;
  const int wm = wave & 3, wn = wave >> 2;

  v8f accg[2][4], accu[2][4];
#pragma unroll
  for (int i = 0; i < 2; ++i)
#pragma unroll
    for (int j = 0; j < 4; ++j)
#pragma unroll
      for (int r = 0; r < 8; ++r) { accg[i][j][r] = 0.f; accu[i][j][r] = 0.f; }

  const int arow = tid >> 1;
  const int akh  = (tid & 1) * 16;
  int grow; float ascale;
  if (GATHER) {
    const bool av = (m0 + arow) < M;
    const int slot = e * T_TOK + m0 + arow;
    grow   = av ? slot_token[slot] : 0;
    ascale = av ? slot_w[slot] : 0.f;
  } else { grow = m0 + arow; ascale = 1.f; }
  const float* ap = x + (size_t)grow * HDIM + akh;

  const int gn = n0 + arow;
  const bool bval = gn < IDIM;
  const unsigned short* gp = wg + (size_t)(bval ? gn : 0) * HDIM + akh;
  const unsigned short* up = wu + (size_t)(bval ? gn : 0) * HDIM + akh;

  // Per-lane LDS destinations (redirect invalid lanes to trash slot)
  const unsigned trash = lds_off32(&lds_trash[tid & 63]);
  unsigned gdst0[2], gdst1[2], udst0[2], udst1[2];
#pragma unroll
  for (int b = 0; b < 2; ++b) {
    gdst0[b] = bval ? lds_off32(&lds_g[b][arow][akh])     : trash;
    gdst1[b] = bval ? lds_off32(&lds_g[b][arow][akh + 8]) : trash;
    udst0[b] = bval ? lds_off32(&lds_u[b][arow][akh])     : trash;
    udst1[b] = bval ? lds_off32(&lds_u[b][arow][akh + 8]) : trash;
  }
  if (!bval) {   // zero never-written rows once (both buffers)
    const uint4 z = make_uint4(0, 0, 0, 0);
#pragma unroll
    for (int b = 0; b < 2; ++b) {
      *(uint4*)&lds_g[b][arow][akh] = z; *(uint4*)&lds_g[b][arow][akh + 8] = z;
      *(uint4*)&lds_u[b][arow][akh] = z; *(uint4*)&lds_u[b][arow][akh + 8] = z;
    }
  }

  auto issueB = [&](int k0, int buf) {
    async_b128(gdst0[buf], gp + k0);
    async_b128(gdst1[buf], gp + k0 + 8);
    async_b128(udst0[buf], up + k0);
    async_b128(udst1[buf], up + k0 + 8);
  };
  auto loadA = [&](int k0, uint4& pa0, uint4& pa1) {
    const float4 a0 = *(const float4*)(ap + k0);
    const float4 a1 = *(const float4*)(ap + k0 + 4);
    const float4 a2 = *(const float4*)(ap + k0 + 8);
    const float4 a3 = *(const float4*)(ap + k0 + 12);
    pa0 = make_uint4(pk2(a0.x * ascale, a0.y * ascale), pk2(a0.z * ascale, a0.w * ascale),
                     pk2(a1.x * ascale, a1.y * ascale), pk2(a1.z * ascale, a1.w * ascale));
    pa1 = make_uint4(pk2(a2.x * ascale, a2.y * ascale), pk2(a2.z * ascale, a2.w * ascale),
                     pk2(a3.x * ascale, a3.y * ascale), pk2(a3.z * ascale, a3.w * ascale));
  };
  auto storeA = [&](int buf, uint4 pa0, uint4 pa1) {
    *(uint4*)&lds_a[buf][arow][akh]     = pa0;
    *(uint4*)&lds_a[buf][arow][akh + 8] = pa1;
  };

  const int KT = HDIM / 32;
  {
    issueB(0, 0);
    uint4 p0, p1; loadA(0, p0, p1); storeA(0, p0, p1);
  }
  for (int kt = 0; kt < KT; ++kt) {
    const int cur = kt & 1, nxt = cur ^ 1;
    const bool more = (kt + 1) < KT;
    uint4 na0, na1;
    if (more) {
      issueB((kt + 1) * 32, nxt);
      loadA((kt + 1) * 32, na0, na1);
      wait_async_le4();          // previous tile's 4 async ops complete (in-order)
    } else {
      wait_async_0();
    }
    __syncthreads();

    const int fr = lane & 15;
    const int ah = (lane >> 4) * 16;   // A: contiguous K block [ah..ah+15]
    const int bh = (lane >> 4) * 16;   // B: contiguous K block [bh..bh+15]
    v16bf afr[2];
#pragma unroll
    for (int i = 0; i < 2; ++i) {
      FragU f;
      const int r = wm * 32 + i * 16 + fr;
      f.q[0] = *(const uint4*)&lds_a[cur][r][ah];
      f.q[1] = *(const uint4*)&lds_a[cur][r][ah + 8];
      afr[i] = f.v;
    }
    v16bf gfr[4], ufr[4];
#pragma unroll
    for (int j = 0; j < 4; ++j) {
      const int r = wn * 64 + j * 16 + fr;
      FragU f1; f1.q[0] = *(const uint4*)&lds_g[cur][r][bh]; f1.q[1] = *(const uint4*)&lds_g[cur][r][bh + 8]; gfr[j] = f1.v;
      FragU f2; f2.q[0] = *(const uint4*)&lds_u[cur][r][bh]; f2.q[1] = *(const uint4*)&lds_u[cur][r][bh + 8]; ufr[j] = f2.v;
    }
#pragma unroll
    for (int i = 0; i < 2; ++i)
#pragma unroll
      for (int j = 0; j < 4; ++j) {
        accg[i][j] = __builtin_amdgcn_wmma_f32_16x16x32_bf16(false, afr[i], false, gfr[j],
                                                             (short)0, accg[i][j], false, false);
        accu[i][j] = __builtin_amdgcn_wmma_f32_16x16x32_bf16(false, afr[i], false, ufr[j],
                                                             (short)0, accu[i][j], false, false);
      }
    __syncthreads();
    if (more) storeA(nxt, na0, na1);
  }

  const int cn = lane & 15;
  const int mh = (lane >> 4) * 8;
#pragma unroll
  for (int i = 0; i < 2; ++i)
#pragma unroll
    for (int j = 0; j < 4; ++j)
#pragma unroll
      for (int r = 0; r < 8; ++r) {
        const int m = m0 + wm * 32 + i * 16 + mh + r;
        const int n = n0 + wn * 64 + j * 16 + cn;
        if (m < M && n < IDIM) {
          const float g = accg[i][j][r];
          const float u = accu[i][j][r];
          const float sv = g / (1.f + __expf(-g));   // SiLU
          act[((size_t)(e * T_TOK) + m) * IDIM + n] = f2bf(sv * u);
        }
      }
}

// ----------------------------------------------------------------- down GEMM
// C = A Wd^T, A = act rows (bf16), K = IDIM, N = HDIM
template <bool SCATTER>
__global__ __launch_bounds__(256)
void moe_down_gemm(const unsigned short* __restrict__ act,
                   const unsigned short* __restrict__ wd_,
                   unsigned long long westride,
                   float* __restrict__ outp,        // direct [T][H] or routed [2][T][H]
                   const int* __restrict__ slot_token,
                   const int* __restrict__ slot_k,
                   const int* __restrict__ counts) {
  const int e  = SCATTER ? (int)blockIdx.z : 0;
  const int M  = SCATTER ? counts[e] : T_TOK;
  const int m0 = blockIdx.y * 128;
  if (m0 >= M) return;
  const int n0 = blockIdx.x * 128;
  const unsigned short* wd = wd_ + (size_t)e * westride;

  __shared__ unsigned short lds_a[2][128][32];
  __shared__ unsigned short lds_b[2][128][32];
  __shared__ uint4 lds_trash[64];

  const int tid = threadIdx.x, lane = tid & 31, wave = tid >> 5;
  const int wm = wave & 3, wn = wave >> 2;

  v8f acc[2][4];
#pragma unroll
  for (int i = 0; i < 2; ++i)
#pragma unroll
    for (int j = 0; j < 4; ++j)
#pragma unroll
      for (int r = 0; r < 8; ++r) acc[i][j][r] = 0.f;

  const int arow = tid >> 1;
  const int akh  = (tid & 1) * 16;
  const bool aval = (m0 + arow) < M;
  const unsigned short* ap =
      act + ((size_t)(e * T_TOK) + (aval ? (m0 + arow) : 0)) * IDIM + akh;
  const int gn = n0 + arow;
  const bool bval = gn < HDIM;
  const unsigned short* bp = wd + (size_t)(bval ? gn : 0) * IDIM + akh;

  const unsigned trash = lds_off32(&lds_trash[tid & 63]);
  unsigned adst0[2], adst1[2], bdst0[2], bdst1[2];
#pragma unroll
  for (int b = 0; b < 2; ++b) {
    adst0[b] = aval ? lds_off32(&lds_a[b][arow][akh])     : trash;
    adst1[b] = aval ? lds_off32(&lds_a[b][arow][akh + 8]) : trash;
    bdst0[b] = bval ? lds_off32(&lds_b[b][arow][akh])     : trash;
    bdst1[b] = bval ? lds_off32(&lds_b[b][arow][akh + 8]) : trash;
  }
  if (!aval || !bval) {
    const uint4 z = make_uint4(0, 0, 0, 0);
#pragma unroll
    for (int b = 0; b < 2; ++b) {
      if (!aval) { *(uint4*)&lds_a[b][arow][akh] = z; *(uint4*)&lds_a[b][arow][akh + 8] = z; }
      if (!bval) { *(uint4*)&lds_b[b][arow][akh] = z; *(uint4*)&lds_b[b][arow][akh + 8] = z; }
    }
  }

  auto issueAB = [&](int k0, int buf) {
    async_b128(adst0[buf], ap + k0);
    async_b128(adst1[buf], ap + k0 + 8);
    async_b128(bdst0[buf], bp + k0);
    async_b128(bdst1[buf], bp + k0 + 8);
  };

  const int KT = IDIM / 32;
  issueAB(0, 0);
  for (int kt = 0; kt < KT; ++kt) {
    const int cur = kt & 1, nxt = cur ^ 1;
    const bool more = (kt + 1) < KT;
    if (more) {
      issueAB((kt + 1) * 32, nxt);
      wait_async_le4();
    } else {
      wait_async_0();
    }
    __syncthreads();

    const int fr = lane & 15;
    const int ah = (lane >> 4) * 16;
    const int bh = (lane >> 4) * 16;
    v16bf afr[2], bfr[4];
#pragma unroll
    for (int i = 0; i < 2; ++i) {
      FragU f;
      const int r = wm * 32 + i * 16 + fr;
      f.q[0] = *(const uint4*)&lds_a[cur][r][ah];
      f.q[1] = *(const uint4*)&lds_a[cur][r][ah + 8];
      afr[i] = f.v;
    }
#pragma unroll
    for (int j = 0; j < 4; ++j) {
      FragU f;
      const int r = wn * 64 + j * 16 + fr;
      f.q[0] = *(const uint4*)&lds_b[cur][r][bh];
      f.q[1] = *(const uint4*)&lds_b[cur][r][bh + 8];
      bfr[j] = f.v;
    }
#pragma unroll
    for (int i = 0; i < 2; ++i)
#pragma unroll
      for (int j = 0; j < 4; ++j)
        acc[i][j] = __builtin_amdgcn_wmma_f32_16x16x32_bf16(false, afr[i], false, bfr[j],
                                                            (short)0, acc[i][j], false, false);
    __syncthreads();
  }

  const int cn = lane & 15;
  const int mh = (lane >> 4) * 8;
#pragma unroll
  for (int i = 0; i < 2; ++i)
#pragma unroll
    for (int j = 0; j < 4; ++j)
#pragma unroll
      for (int r = 0; r < 8; ++r) {
        const int m = m0 + wm * 32 + i * 16 + mh + r;
        const int n = n0 + wn * 64 + j * 16 + cn;
        if (m < M && n < HDIM) {
          const float v = acc[i][j][r];
          if (SCATTER) {
            const int slot = e * T_TOK + m;
            const int tok  = slot_token[slot];
            const int kk   = slot_k[slot];
            outp[((size_t)kk * T_TOK + tok) * HDIM + n] = v;
          } else {
            outp[(size_t)m * HDIM + n] = v;
          }
        }
      }
}

// ----------------------------------------------------------------- final add
__global__ void moe_final_add(float* __restrict__ out, const float* __restrict__ rb, int n) {
  const int i = blockIdx.x * blockDim.x + threadIdx.x;
  if (i < n) out[i] += rb[i] + rb[(size_t)n + i];
}

// ------------------------------------------------------------------- launch
extern "C" void kernel_launch(void* const* d_in, const int* in_sizes, int n_in,
                              void* d_out, int out_size, void* d_ws, size_t ws_size,
                              hipStream_t stream) {
  const float* x   = (const float*)d_in[0];
  const float* rw  = (const float*)d_in[1];
  const float* gup = (const float*)d_in[2];
  const float* dwn = (const float*)d_in[3];
  const float* sgw = (const float*)d_in[4];
  const float* suw = (const float*)d_in[5];
  const float* sdw = (const float*)d_in[6];

  float* out    = (float*)d_out;                  // [T][H]
  float* scores = out + (size_t)T_TOK * HDIM;     // [E][T]

  char* ws = (char*)d_ws;
  size_t off = 0;
  auto take = [&](size_t bytes) -> void* {
    void* p = ws + off;
    off += (bytes + 255) & ~(size_t)255;
    return p;
  };
  unsigned short* gup_t  = (unsigned short*)take((size_t)NEXP * 2 * IDIM * HDIM * 2);
  unsigned short* down_t = (unsigned short*)take((size_t)NEXP * HDIM * IDIM * 2);
  unsigned short* sg_bf  = (unsigned short*)take((size_t)IDIM * HDIM * 2);
  unsigned short* su_bf  = (unsigned short*)take((size_t)IDIM * HDIM * 2);
  unsigned short* sd_bf  = (unsigned short*)take((size_t)HDIM * IDIM * 2);
  unsigned short* act_sh = (unsigned short*)take((size_t)T_TOK * IDIM * 2);
  unsigned short* act_rt = (unsigned short*)take((size_t)NEXP * T_TOK * IDIM * 2);
  float* routed_buf      = (float*)take((size_t)2 * T_TOK * HDIM * 4);
  int*   slot_token      = (int*)take((size_t)NEXP * T_TOK * 4);
  float* slot_w          = (float*)take((size_t)NEXP * T_TOK * 4);
  int*   slot_k          = (int*)take((size_t)NEXP * T_TOK * 4);
  int*   counts          = (int*)take(256);

  moe_zero_counts<<<1, 64, 0, stream>>>(counts);
  moe_router_kernel<<<T_TOK / 256, 256, 0, stream>>>(x, rw, scores,
                                                     slot_token, slot_w, slot_k, counts);

  const int nw = IDIM * HDIM;
  moe_conv_kernel<<<(nw + 255) / 256, 256, 0, stream>>>(sgw, sg_bf, nw);
  moe_conv_kernel<<<(nw + 255) / 256, 256, 0, stream>>>(suw, su_bf, nw);
  moe_conv_kernel<<<(nw + 255) / 256, 256, 0, stream>>>(sdw, sd_bf, nw);

  // gate_up_proj [E,H,2I] -> [E,2I,H] bf16 ; down_proj [E,I,H] -> [E,H,I] bf16
  moe_transpose_conv<<<dim3(2 * IDIM / 32, HDIM / 32, NEXP), dim3(32, 8), 0, stream>>>(
      gup, gup_t, HDIM, 2 * IDIM);
  moe_transpose_conv<<<dim3(HDIM / 32, IDIM / 32, NEXP), dim3(32, 8), 0, stream>>>(
      dwn, down_t, IDIM, HDIM);

  const dim3 blk(256);
  // shared gate/up  (M=T, N=I, K=H)
  moe_gateup_gemm<false><<<dim3((IDIM + 127) / 128, T_TOK / 128, 1), blk, 0, stream>>>(
      x, sg_bf, su_bf, 0ull, act_sh, nullptr, nullptr, nullptr);
  // routed gate/up  (per expert, gathered+scaled rows of x)
  moe_gateup_gemm<true><<<dim3((IDIM + 127) / 128, T_TOK / 128, NEXP), blk, 0, stream>>>(
      x, gup_t, gup_t + (size_t)IDIM * HDIM,
      (unsigned long long)(2 * (size_t)IDIM * HDIM), act_rt, slot_token, slot_w, counts);
  // shared down -> writes d_out directly (full coverage)
  moe_down_gemm<false><<<dim3(HDIM / 128, T_TOK / 128, 1), blk, 0, stream>>>(
      act_sh, sd_bf, 0ull, out, nullptr, nullptr, nullptr);
  // routed down -> scatter into routed_buf[2][T][H]
  moe_down_gemm<true><<<dim3(HDIM / 128, T_TOK / 128, NEXP), blk, 0, stream>>>(
      act_rt, down_t, (unsigned long long)((size_t)HDIM * IDIM), routed_buf,
      slot_token, slot_k, counts);

  moe_final_add<<<(T_TOK * HDIM) / 256, 256, 0, stream>>>(out, routed_buf, T_TOK * HDIM);

  (void)in_sizes; (void)n_in; (void)out_size; (void)ws_size;
}
